// LoRAConv1D_6158983102942
// MI455X (gfx1250) — matrix operand order repaired and checked by reading, not compiled, not run on hardware
//
#include <hip/hip_runtime.h>

// LoRA Conv1D:  out = x @ W + b + SCALING * (x @ A) @ B
// M = 8192, K = 1600, N = 4800, R = 8, SCALING = 4.0
//
// Compute-bound GEMM on the bf16 matrix pipes: fp32 operands split once into
// bf16 hi/lo planes (x row-major, W transposed to [N][K]) in workspace, then
// the GEMM accumulates ahi*bhi + ahi*blo + alo*bhi with
// V_WMMA_F32_16X16X32_BF16 (~1e-5 relative accuracy, K=32 per matrix op).
// Rank-8 LoRA tail stays exact fp32 WMMA 16x16x4.

#define M_DIM 8192
#define K_DIM 1600
#define N_DIM 4800
#define R_DIM 8
#define SCALING 4.0f

#define BM 128   // workgroup M tile
#define BN 96    // workgroup N tile  (4800 / 96 = 50, exact)
#define BK 64    // K slice per stage (1600 / 64 = 25, exact)
#define LDA 72   // LDS row stride in bf16 units (64 + 8 pad; 144B, 16B-aligned)
#define LDTF 12  // LDS row stride in floats for the fp32 LoRA tail

typedef __attribute__((ext_vector_type(2)))  float  v2f;
typedef __attribute__((ext_vector_type(8)))  float  v8f;
typedef __attribute__((ext_vector_type(8)))  __bf16 v8bf;
typedef __attribute__((ext_vector_type(16))) __bf16 v16bf;
typedef unsigned short ushort_t;

// ---- bf16 split helpers (round-to-nearest-even) ----
__device__ __forceinline__ unsigned short bf_rne(float f) {
  unsigned u = __builtin_bit_cast(unsigned int, f);
  return (unsigned short)((u + 0x7FFFu + ((u >> 16) & 1u)) >> 16);
}
__device__ __forceinline__ float bf_tof(unsigned short h) {
  return __builtin_bit_cast(float, ((unsigned int)h) << 16);
}
__device__ __forceinline__ void split2(float v, unsigned short& h,
                                       unsigned short& l) {
  h = bf_rne(v);
  l = bf_rne(v - bf_tof(h));
}
__device__ __forceinline__ unsigned int pk2(unsigned short a,
                                            unsigned short b) {
  return (unsigned int)a | ((unsigned int)b << 16);
}
// 32B fragment = two 16B-aligned LDS reads, concatenated
__device__ __forceinline__ v16bf ldfrag(const unsigned short* p) {
  v8bf a = *(const v8bf*)p;
  v8bf b = *(const v8bf*)(p + 8);
  return __builtin_shufflevector(a, b, 0, 1, 2, 3, 4, 5, 6, 7, 8, 9, 10, 11,
                                 12, 13, 14, 15);
}

// ---------------------------------------------------------------------------
// Prep A: t[m][r] = SCALING * sum_k x[m][k] * loraA[k][r]   (tiny, exact)
// ---------------------------------------------------------------------------
__global__ __launch_bounds__(256) void lora_xa_kernel(
    const float* __restrict__ x, const float* __restrict__ loraA,
    float* __restrict__ t) {
  __shared__ float Alds[K_DIM * R_DIM];  // 51.2 KB
  const int tid = threadIdx.x;
  for (int i = tid; i < K_DIM * R_DIM; i += 256) Alds[i] = loraA[i];
  __syncthreads();

  const int m = blockIdx.x * 32 + (tid >> 3);
  const int r = tid & 7;
  const float4* xrow = (const float4*)(x + (size_t)m * K_DIM);
  float acc = 0.f;
  for (int k4 = 0; k4 < K_DIM / 4; ++k4) {
    float4 xv = xrow[k4];
    const int kb = k4 * 4;
    acc += xv.x * Alds[(kb + 0) * 8 + r];
    acc += xv.y * Alds[(kb + 1) * 8 + r];
    acc += xv.z * Alds[(kb + 2) * 8 + r];
    acc += xv.w * Alds[(kb + 3) * 8 + r];
  }
  t[(size_t)m * 8 + r] = SCALING * acc;
}

// ---------------------------------------------------------------------------
// Prep B: split x (row-major) into bf16 hi/lo planes, one float4 per thread
// ---------------------------------------------------------------------------
__global__ __launch_bounds__(256) void split_x_kernel(
    const float* __restrict__ x, ushort_t* __restrict__ xh,
    ushort_t* __restrict__ xl) {
  const size_t idx = (size_t)blockIdx.x * 256 + threadIdx.x;  // float4 index
  float4 v = ((const float4*)x)[idx];
  unsigned short h0, h1, h2, h3, l0, l1, l2, l3;
  split2(v.x, h0, l0);
  split2(v.y, h1, l1);
  split2(v.z, h2, l2);
  split2(v.w, h3, l3);
  *(uint2*)(xh + idx * 4) = make_uint2(pk2(h0, h1), pk2(h2, h3));
  *(uint2*)(xl + idx * 4) = make_uint2(pk2(l0, l1), pk2(l2, l3));
}

// ---------------------------------------------------------------------------
// Prep C: split W and transpose to [N][K] bf16 hi/lo planes (64x64 LDS tiles)
// ---------------------------------------------------------------------------
#define TS 72  // tile LDS stride (bf16 units)
__global__ __launch_bounds__(256) void split_wT_kernel(
    const float* __restrict__ W, ushort_t* __restrict__ whT,
    ushort_t* __restrict__ wlT) {
  __shared__ __align__(16) ushort_t Th[64 * TS];
  __shared__ __align__(16) ushort_t Tl[64 * TS];
  const int tid = threadIdx.x;
  const int n0 = blockIdx.x * 64;  // 4800/64 = 75
  const int k0 = blockIdx.y * 64;  // 1600/64 = 25

  // coalesced fp32 reads along N; split; transposed scalar stores into LDS
#pragma unroll
  for (int i = 0; i < 4; ++i) {
    const int f = tid + i * 256, kk = f >> 4, n4 = f & 15;
    float4 v = *(const float4*)(W + (size_t)(k0 + kk) * N_DIM + n0 + n4 * 4);
    const float e[4] = {v.x, v.y, v.z, v.w};
#pragma unroll
    for (int j = 0; j < 4; ++j) {
      unsigned short h, l;
      split2(e[j], h, l);
      Th[(n4 * 4 + j) * TS + kk] = h;
      Tl[(n4 * 4 + j) * TS + kk] = l;
    }
  }
  __syncthreads();
  // coalesced bf16 writes along K
#pragma unroll
  for (int i = 0; i < 2; ++i) {
    const int f = tid + i * 256, n = f >> 3, c8 = f & 7;
    *(uint4*)(whT + (size_t)(n0 + n) * K_DIM + k0 + c8 * 8) =
        *(const uint4*)(&Th[n * TS + c8 * 8]);
    *(uint4*)(wlT + (size_t)(n0 + n) * K_DIM + k0 + c8 * 8) =
        *(const uint4*)(&Tl[n * TS + c8 * 8]);
  }
}

// ---------------------------------------------------------------------------
// Main GEMM (pre-split operands): out = x @ W + bias + t @ loraB
//   8 waves; WG tile 128x96; wave tile 32x48 (2x3 16x16 frags)
// ---------------------------------------------------------------------------
__global__ __launch_bounds__(256) void lora_gemm_kernel(
    const ushort_t* __restrict__ xh, const ushort_t* __restrict__ xl,
    const ushort_t* __restrict__ whT, const ushort_t* __restrict__ wlT,
    const float* __restrict__ bias, const float* __restrict__ loraB,
    const float* __restrict__ t, float* __restrict__ out) {
  __shared__ __align__(16) unsigned short smem[(2 * BM + 2 * BN) * LDA];
  unsigned short* Ah = smem;
  unsigned short* Al = Ah + BM * LDA;
  unsigned short* Bh = Al + BM * LDA;
  unsigned short* Bl = Bh + BN * LDA;
  float* Atf = (float*)smem;                   // fp32 alias for LoRA tail
  float* Btf = (float*)(smem + 2 * BM * LDA);

  const int tid  = threadIdx.x;
  const int lane = tid & 31;
  const int wave = tid >> 5;
  const int wm   = wave & 3;   // 4 waves along M -> 32 rows each
  const int wn   = wave >> 2;  // 2 waves along N -> 48 cols each

  const int m_blk = blockIdx.y * BM;
  const int n_blk = blockIdx.x * BN;

  const int lrow = lane & 15;
  const int lhi  = lane >> 4;

  v8f c[2][3];
  {
    v8f z = {};
#pragma unroll
    for (int mi = 0; mi < 2; ++mi)
#pragma unroll
      for (int ni = 0; ni < 3; ++ni) c[mi][ni] = z;
  }

  for (int kt = 0; kt < K_DIM / BK; ++kt) {
    const int k0 = kt * BK;

    // ---- global -> regs: 16B bf16 chunks (8 elements along K) ----
    uint4 arh[4], arl[4];  // A: 128 rows x 8 chunks = 1024 / 256 thr
#pragma unroll
    for (int i = 0; i < 4; ++i) {
      const int f = tid + i * 256, row = f >> 3, c8 = f & 7;
      const size_t g = (size_t)(m_blk + row) * K_DIM + k0 + c8 * 8;
      arh[i] = *(const uint4*)(xh + g);
      arl[i] = *(const uint4*)(xl + g);
    }
    uint4 brh[3], brl[3];  // B: 96 rows x 8 chunks = 768 / 256 thr
#pragma unroll
    for (int i = 0; i < 3; ++i) {
      const int f = tid + i * 256, n = f >> 3, c8 = f & 7;
      const size_t g = (size_t)(n_blk + n) * K_DIM + k0 + c8 * 8;
      brh[i] = *(const uint4*)(whT + g);
      brl[i] = *(const uint4*)(wlT + g);
    }

    __syncthreads();  // previous stage's LDS reads complete

    // ---- regs -> LDS, all 16B stores ----
    // A: natural K order (same as B) so a lane's 16x16x32 fragment
    //    (contiguous k = 16h + s) is 32 contiguous bytes.
#pragma unroll
    for (int i = 0; i < 4; ++i) {
      const int f = tid + i * 256, row = f >> 3, c8 = f & 7;
      const int base = row * LDA + c8 * 8;
      *(uint4*)(&Ah[base]) = arh[i];
      *(uint4*)(&Al[base]) = arl[i];
    }
#pragma unroll
    for (int i = 0; i < 3; ++i) {  // B: natural K order
      const int f = tid + i * 256, n = f >> 3, c8 = f & 7;
      const int base = n * LDA + c8 * 8;
      *(uint4*)(&Bh[base]) = brh[i];
      *(uint4*)(&Bl[base]) = brl[i];
    }
    __syncthreads();

    // ---- 2 x (16x16x32) k-chunks; 3 bf16 WMMAs per C frag per chunk ----
#pragma unroll
    for (int kc = 0; kc < BK / 32; ++kc) {
      v16bf ah[2], al[2], bh[3], bl[3];
#pragma unroll
      for (int mi = 0; mi < 2; ++mi) {
        const int off = (wm * 32 + mi * 16 + lrow) * LDA + kc * 32 + 16 * lhi;
        ah[mi] = ldfrag(&Ah[off]);
        al[mi] = ldfrag(&Al[off]);
      }
#pragma unroll
      for (int ni = 0; ni < 3; ++ni) {
        const int off = (wn * 48 + ni * 16 + lrow) * LDA + kc * 32 + 16 * lhi;
        bh[ni] = ldfrag(&Bh[off]);
        bl[ni] = ldfrag(&Bl[off]);
      }
#pragma unroll
      for (int mi = 0; mi < 2; ++mi)
#pragma unroll
        for (int ni = 0; ni < 3; ++ni) {
          c[mi][ni] = __builtin_amdgcn_wmma_f32_16x16x32_bf16(
              false, ah[mi], false, bh[ni], (short)0, c[mi][ni], false, false);
          c[mi][ni] = __builtin_amdgcn_wmma_f32_16x16x32_bf16(
              false, ah[mi], false, bl[ni], (short)0, c[mi][ni], false, false);
          c[mi][ni] = __builtin_amdgcn_wmma_f32_16x16x32_bf16(
              false, al[mi], false, bh[ni], (short)0, c[mi][ni], false, false);
        }
    }
  }

  // ---- LoRA rank-8 tail: exact fp32 WMMA 16x16x4, two k4-steps ----
  __syncthreads();
#pragma unroll
  for (int i = 0; i < 4; ++i) {  // t tile: 128x8
    const int f = tid + i * 256, row = f >> 3, rr = f & 7;
    Atf[row * LDTF + rr] = t[(size_t)(m_blk + row) * 8 + rr];
  }
#pragma unroll
  for (int i = 0; i < 3; ++i) {  // loraB^T tile: 96x8
    const int f = tid + i * 256, n = f >> 3, rr = f & 7;
    Btf[n * LDTF + rr] = loraB[(size_t)rr * N_DIM + n_blk + n];
  }
  __syncthreads();
#pragma unroll
  for (int kk = 0; kk < 2; ++kk) {
    const int kf = kk * 4 + lhi * 2;
    v2f a[2], b[3];
#pragma unroll
    for (int mi = 0; mi < 2; ++mi)
      a[mi] = *(const v2f*)(&Atf[(wm * 32 + mi * 16 + lrow) * LDTF + kf]);
#pragma unroll
    for (int ni = 0; ni < 3; ++ni)
      b[ni] = *(const v2f*)(&Btf[(wn * 48 + ni * 16 + lrow) * LDTF + kf]);
#pragma unroll
    for (int mi = 0; mi < 2; ++mi)
#pragma unroll
      for (int ni = 0; ni < 3; ++ni)
        c[mi][ni] = __builtin_amdgcn_wmma_f32_16x16x4_f32(
            false, a[mi], false, b[ni], (short)0, c[mi][ni], false, false);
  }

  // ---- epilogue: add bias, store.  C layout: VGPR r -> M = r + 8*lhi ----
#pragma unroll
  for (int ni = 0; ni < 3; ++ni) {
    const int n = n_blk + wn * 48 + ni * 16 + lrow;
    const float bv = bias[n];
#pragma unroll
    for (int mi = 0; mi < 2; ++mi) {
      const int mbase = m_blk + wm * 32 + mi * 16 + lhi * 8;
#pragma unroll
      for (int r = 0; r < 8; ++r)
        out[(size_t)(mbase + r) * N_DIM + n] = c[mi][ni][r] + bv;
    }
  }
}

// ---------------------------------------------------------------------------
// Fallback GEMM (split on the fly in LDS) if workspace is too small for the
// pre-split planes. Same tiling and WMMA mix as the fast path.
// ---------------------------------------------------------------------------
__global__ __launch_bounds__(256) void lora_gemm_fallback_kernel(
    const float* __restrict__ x, const float* __restrict__ W,
    const float* __restrict__ bias, const float* __restrict__ loraB,
    const float* __restrict__ t, float* __restrict__ out) {
  __shared__ __align__(16) unsigned short smem[(2 * BM + 2 * BN) * LDA];
  unsigned short* Ah = smem;
  unsigned short* Al = Ah + BM * LDA;
  unsigned short* Bh = Al + BM * LDA;
  unsigned short* Bl = Bh + BN * LDA;
  float* Atf = (float*)smem;
  float* Btf = (float*)(smem + 2 * BM * LDA);

  const int tid  = threadIdx.x;
  const int lane = tid & 31;
  const int wave = tid >> 5;
  const int wm   = wave & 3;
  const int wn   = wave >> 2;
  const int m_blk = blockIdx.y * BM;
  const int n_blk = blockIdx.x * BN;
  const int lrow = lane & 15;
  const int lhi  = lane >> 4;

  v8f c[2][3];
  {
    v8f z = {};
#pragma unroll
    for (int mi = 0; mi < 2; ++mi)
#pragma unroll
      for (int ni = 0; ni < 3; ++ni) c[mi][ni] = z;
  }

  for (int kt = 0; kt < K_DIM / BK; ++kt) {
    const int k0 = kt * BK;
    float4 ar[8];
#pragma unroll
    for (int i = 0; i < 8; ++i) {
      const int f = tid + i * 256, row = f >> 4, c4 = f & 15;
      ar[i] = *(const float4*)(x + (size_t)(m_blk + row) * K_DIM + k0 + c4 * 4);
    }
    float4 br[6];
#pragma unroll
    for (int i = 0; i < 6; ++i) {
      const int f = tid + i * 256, kk = f / 24, n4 = f % 24;
      br[i] = *(const float4*)(W + (size_t)(k0 + kk) * N_DIM + n_blk + n4 * 4);
    }
    __syncthreads();
#pragma unroll
    for (int i = 0; i < 8; ++i) {
      const int f = tid + i * 256, row = f >> 4, c4 = f & 15;
      const int k = c4 * 4;
      const int base = row * LDA + k;
      unsigned short h0, h1, h2, h3, l0, l1, l2, l3;
      split2(ar[i].x, h0, l0);
      split2(ar[i].y, h1, l1);
      split2(ar[i].z, h2, l2);
      split2(ar[i].w, h3, l3);
      *(uint2*)(&Ah[base]) = make_uint2(pk2(h0, h1), pk2(h2, h3));
      *(uint2*)(&Al[base]) = make_uint2(pk2(l0, l1), pk2(l2, l3));
    }
#pragma unroll
    for (int i = 0; i < 6; ++i) {
      const int f = tid + i * 256, kk = f / 24, n4 = f % 24;
      const float v[4] = {br[i].x, br[i].y, br[i].z, br[i].w};
#pragma unroll
      for (int j = 0; j < 4; ++j) {
        unsigned short h, l;
        split2(v[j], h, l);
        Bh[(n4 * 4 + j) * LDA + kk] = h;
        Bl[(n4 * 4 + j) * LDA + kk] = l;
      }
    }
    __syncthreads();
#pragma unroll
    for (int kc = 0; kc < BK / 32; ++kc) {
      v16bf ah[2], al[2], bh[3], bl[3];
#pragma unroll
      for (int mi = 0; mi < 2; ++mi) {
        const int off = (wm * 32 + mi * 16 + lrow) * LDA + kc * 32 + 16 * lhi;
        ah[mi] = ldfrag(&Ah[off]);
        al[mi] = ldfrag(&Al[off]);
      }
#pragma unroll
      for (int ni = 0; ni < 3; ++ni) {
        const int off = (wn * 48 + ni * 16 + lrow) * LDA + kc * 32 + 16 * lhi;
        bh[ni] = ldfrag(&Bh[off]);
        bl[ni] = ldfrag(&Bl[off]);
      }
#pragma unroll
      for (int mi = 0; mi < 2; ++mi)
#pragma unroll
        for (int ni = 0; ni < 3; ++ni) {
          c[mi][ni] = __builtin_amdgcn_wmma_f32_16x16x32_bf16(
              false, ah[mi], false, bh[ni], (short)0, c[mi][ni], false, false);
          c[mi][ni] = __builtin_amdgcn_wmma_f32_16x16x32_bf16(
              false, ah[mi], false, bl[ni], (short)0, c[mi][ni], false, false);
          c[mi][ni] = __builtin_amdgcn_wmma_f32_16x16x32_bf16(
              false, al[mi], false, bh[ni], (short)0, c[mi][ni], false, false);
        }
    }
  }

  __syncthreads();
#pragma unroll
  for (int i = 0; i < 4; ++i) {
    const int f = tid + i * 256, row = f >> 3, rr = f & 7;
    Atf[row * LDTF + rr] = t[(size_t)(m_blk + row) * 8 + rr];
  }
#pragma unroll
  for (int i = 0; i < 3; ++i) {
    const int f = tid + i * 256, n = f >> 3, rr = f & 7;
    Btf[n * LDTF + rr] = loraB[(size_t)rr * N_DIM + n_blk + n];
  }
  __syncthreads();
#pragma unroll
  for (int kk = 0; kk < 2; ++kk) {
    const int kf = kk * 4 + lhi * 2;
    v2f a[2], b[3];
#pragma unroll
    for (int mi = 0; mi < 2; ++mi)
      a[mi] = *(const v2f*)(&Atf[(wm * 32 + mi * 16 + lrow) * LDTF + kf]);
#pragma unroll
    for (int ni = 0; ni < 3; ++ni)
      b[ni] = *(const v2f*)(&Btf[(wn * 48 + ni * 16 + lrow) * LDTF + kf]);
#pragma unroll
    for (int mi = 0; mi < 2; ++mi)
#pragma unroll
      for (int ni = 0; ni < 3; ++ni)
        c[mi][ni] = __builtin_amdgcn_wmma_f32_16x16x4_f32(
            false, a[mi], false, b[ni], (short)0, c[mi][ni], false, false);
  }

#pragma unroll
  for (int ni = 0; ni < 3; ++ni) {
    const int n = n_blk + wn * 48 + ni * 16 + lrow;
    const float bv = bias[n];
#pragma unroll
    for (int mi = 0; mi < 2; ++mi) {
      const int mbase = m_blk + wm * 32 + mi * 16 + lhi * 8;
#pragma unroll
      for (int r = 0; r < 8; ++r)
        out[(size_t)(mbase + r) * N_DIM + n] = c[mi][ni][r] + bv;
    }
  }
}

// ---------------------------------------------------------------------------
extern "C" void kernel_launch(void* const* d_in, const int* in_sizes, int n_in,
                              void* d_out, int out_size, void* d_ws,
                              size_t ws_size, hipStream_t stream) {
  const float* x     = (const float*)d_in[0];
  const float* W     = (const float*)d_in[1];
  const float* bias  = (const float*)d_in[2];
  const float* loraA = (const float*)d_in[3];
  const float* loraB = (const float*)d_in[4];
  float* out = (float*)d_out;

  // workspace layout (all 256B-aligned offsets)
  const size_t t_bytes  = (size_t)M_DIM * R_DIM * 4;       //   256 KB
  const size_t xp_bytes = (size_t)M_DIM * K_DIM * 2;       //  25 MB per plane
  const size_t wp_bytes = (size_t)N_DIM * K_DIM * 2;       //  14.6 MB per plane
  const size_t need = t_bytes + 2 * xp_bytes + 2 * wp_bytes;

  char* ws = (char*)d_ws;
  float* t = (float*)ws;

  lora_xa_kernel<<<M_DIM / 32, 256, 0, stream>>>(x, loraA, t);

  dim3 grid(N_DIM / BN, M_DIM / BM);  // 50 x 64, exact

  if (ws_size >= need) {
    ushort_t* xh  = (ushort_t*)(ws + t_bytes);
    ushort_t* xl  = (ushort_t*)(ws + t_bytes + xp_bytes);
    ushort_t* whT = (ushort_t*)(ws + t_bytes + 2 * xp_bytes);
    ushort_t* wlT = (ushort_t*)(ws + t_bytes + 2 * xp_bytes + wp_bytes);

    split_x_kernel<<<(M_DIM * K_DIM / 4) / 256, 256, 0, stream>>>(x, xh, xl);
    dim3 tgrid(N_DIM / 64, K_DIM / 64);  // 75 x 25
    split_wT_kernel<<<tgrid, 256, 0, stream>>>(W, whT, wlT);

    lora_gemm_kernel<<<grid, 256, 0, stream>>>(xh, xl, whT, wlT, bias, loraB,
                                               t, out);
  } else {
    lora_gemm_fallback_kernel<<<grid, 256, 0, stream>>>(x, W, bias, loraB, t,
                                                        out);
  }
}
